// QuantizedLinear_13460427506048
// MI455X (gfx1250) — compile-verified, matrix operand order checked by reading, not run on hardware
//
#include <hip/hip_runtime.h>
#include <hip/hip_bf16.h>
#include <stdint.h>

typedef __attribute__((ext_vector_type(16))) _Float16 v16h;
typedef __attribute__((ext_vector_type(8)))  _Float16 v8h;
typedef __attribute__((ext_vector_type(4)))  _Float16 v4h;
typedef __attribute__((ext_vector_type(8)))  float    v8f;
typedef __attribute__((ext_vector_type(4)))  float    v4f;
typedef __attribute__((ext_vector_type(4)))  int      v4i;

// address-space-qualified v4i pointers for the async-to-LDS builtin
typedef __attribute__((address_space(1))) v4i* as1_v4i_p;
typedef __attribute__((address_space(3))) v4i* as3_v4i_p;

#define M_DIM 8192   // B*S = 4*2048
#define N_DIM 4096   // OUT
#define K_DIM 4096   // IN
#define BM 128
#define BN 128
#define BK 32
#define LDS_STRIDE 40   // 32 halfs data + 8 pad -> 80B rows (16B aligned);
                        // dword bank = r*20 mod 64: 16 distinct banks over 16 rows

#if __has_builtin(__builtin_amdgcn_global_load_async_to_lds_b128)
#define HAVE_ASYNC_LDS 1
#else
#define HAVE_ASYNC_LDS 0
#endif

// ---------------- prep kernels (bandwidth-bound) ----------------

__global__ void convert_x_f16(const float* __restrict__ x, _Float16* __restrict__ x16) {
    size_t i = ((size_t)blockIdx.x * blockDim.x + threadIdx.x) * 4;
    v4f v = *(const v4f*)(x + i);
    v4h h;
    h[0] = (_Float16)v[0];
    h[1] = (_Float16)v[1];
    h[2] = (_Float16)v[2];
    h[3] = (_Float16)v[3];
    *(v4h*)(x16 + i) = h;
}

__global__ void dequant_w_f16(const int* __restrict__ wq,
                              const float* __restrict__ absmax,
                              const float* __restrict__ code,
                              _Float16* __restrict__ w16) {
    size_t i = ((size_t)blockIdx.x * blockDim.x + threadIdx.x) * 4;
    v4i q = *(const v4i*)(wq + i);
    float s = absmax[i >> 12];   // BLOCKSIZE = 4096; i%4==0 so all 4 share one block
    v4h h;
    h[0] = (_Float16)(code[q[0] & 255] * s);
    h[1] = (_Float16)(code[q[1] & 255] * s);
    h[2] = (_Float16)(code[q[2] & 255] * s);
    h[3] = (_Float16)(code[q[3] & 255] * s);
    *(v4h*)(w16 + i) = h;
}

// ---------------- shared helpers ----------------

union Frag16 { v16h v; v8h h[2]; };

__device__ __forceinline__ void wait_asynccnt_4() {
#if HAVE_ASYNC_LDS
#if __has_builtin(__builtin_amdgcn_s_wait_asynccnt)
    __builtin_amdgcn_s_wait_asynccnt(4);
#else
    asm volatile("s_wait_asynccnt 0x4" ::: "memory");
#endif
#endif
}
__device__ __forceinline__ void wait_asynccnt_0() {
#if HAVE_ASYNC_LDS
#if __has_builtin(__builtin_amdgcn_s_wait_asynccnt)
    __builtin_amdgcn_s_wait_asynccnt(0);
#else
    asm volatile("s_wait_asynccnt 0x0" ::: "memory");
#endif
#endif
}

#if HAVE_ASYNC_LDS
__device__ __forceinline__ void async_copy_b128(const _Float16* gsrc, _Float16* ldsdst) {
    // AS1 ptr value == generic; AS3 ptr value = low 32 bits of generic LDS addr.
    __builtin_amdgcn_global_load_async_to_lds_b128(
        (as1_v4i_p)(uintptr_t)gsrc,
        (as3_v4i_p)(unsigned)(uintptr_t)ldsdst,
        0, 0);
}
#endif

// ---------------- WMMA GEMM: C[M][N] = A[M][K] * B[N][K]^T + bias ----------------

#if HAVE_ASYNC_LDS
// Triple-buffered async-to-LDS pipeline: prefetch depth 1, ONE barrier/iter.
// Safety: wave skew < 1 iteration, so an issue into buf[(kt+1)%3] can only
// race a laggard reading buf[(kt-1)%3]; distinct since 2 !≡ 0 (mod 3).
// Loop is unroll-by-3 with two peeled tail iterations so every buffer index
// is a compile-time constant (LDS offsets fold into immediates).
__global__ __launch_bounds__(256) void wmma_gemm_f16_async(
    const _Float16* __restrict__ A,   // [M][K] f16
    const _Float16* __restrict__ B,   // [N][K] f16
    const float*    __restrict__ bias,
    float*          __restrict__ C)   // [M][N] f32
{
    __shared__ __align__(16) _Float16 ldsA[3][BM * LDS_STRIDE];   // 30 KB
    __shared__ __align__(16) _Float16 ldsB[3][BN * LDS_STRIDE];   // 30 KB

    const int tid   = threadIdx.x;
    const int lane  = tid & 31;
    const int wave  = tid >> 5;      // 0..7
    const int waveM = wave >> 2;     // 0..1
    const int waveN = wave & 3;      // 0..3

    const int m0 = blockIdx.y * BM;
    const int n0 = blockIdx.x * BN;

    const int lr = tid >> 2;          // 0..63
    const int lc = (tid & 3) * 8;     // 0,8,16,24 halfs

    const _Float16* gA = A + (size_t)(m0 + lr) * K_DIM + lc;
    const _Float16* gB = B + (size_t)(n0 + lr) * K_DIM + lc;

    v8f acc[4][2] = {};

    const int lm = lane & 15;          // row within 16x16 tile
    const int kh = (lane >> 4) * 8;    // K-chunk select per ISA 7.12.2

    constexpr int KT = K_DIM / BK;     // 128 K-steps; KT-2 = 126 = 3*42

    auto issue = [&](int kt, int c) {
        const _Float16* ga = gA + (size_t)kt * BK;
        const _Float16* gb = gB + (size_t)kt * BK;
        async_copy_b128(ga,                     &ldsA[c][lr * LDS_STRIDE + lc]);
        async_copy_b128(ga + (size_t)64*K_DIM,  &ldsA[c][(64+lr) * LDS_STRIDE + lc]);
        async_copy_b128(gb,                     &ldsB[c][lr * LDS_STRIDE + lc]);
        async_copy_b128(gb + (size_t)64*K_DIM,  &ldsB[c][(64+lr) * LDS_STRIDE + lc]);
    };

    auto compute = [&](int c) {
        Frag16 fb[2];
#pragma unroll
        for (int nt = 0; nt < 2; ++nt) {
            const int row = waveN * 32 + nt * 16 + lm;
            fb[nt].h[0] = *(const v8h*)&ldsB[c][row * LDS_STRIDE + kh];
            fb[nt].h[1] = *(const v8h*)&ldsB[c][row * LDS_STRIDE + 16 + kh];
        }
        Frag16 fa[4];
#pragma unroll
        for (int mt = 0; mt < 4; ++mt) {
            const int row = waveM * 64 + mt * 16 + lm;
            fa[mt].h[0] = *(const v8h*)&ldsA[c][row * LDS_STRIDE + kh];
            fa[mt].h[1] = *(const v8h*)&ldsA[c][row * LDS_STRIDE + 16 + kh];
        }
#pragma unroll
        for (int mt = 0; mt < 4; ++mt) {
#pragma unroll
            for (int nt = 0; nt < 2; ++nt) {
                acc[mt][nt] = __builtin_amdgcn_wmma_f32_16x16x32_f16(
                    false, fa[mt].v, false, fb[nt].v,
                    (short)0, acc[mt][nt], false, false);
            }
        }
    };

    // prologue: batch 0 -> buffer 0
    issue(0, 0);

    // steady state: 126 iterations, unrolled by 3 so kt%3 is compile-time
#pragma unroll 3
    for (int kt = 0; kt < KT - 2; ++kt) {
        issue(kt + 1, (kt + 1) % 3);
        wait_asynccnt_4();   // current tile landed; next batch still in flight
        __syncthreads();     // single barrier per iteration
        compute(kt % 3);
    }

    // tail kt = 126: compute buf 0 (126%3), prefetch 127 -> buf 1 (127%3)
    issue(KT - 1, 1);
    wait_asynccnt_4();
    __syncthreads();
    compute(0);

    // tail kt = 127: last tile in buf 1
    wait_asynccnt_0();
    __syncthreads();
    compute(1);

#pragma unroll
    for (int mt = 0; mt < 4; ++mt) {
#pragma unroll
        for (int nt = 0; nt < 2; ++nt) {
            const int n     = n0 + waveN * 32 + nt * 16 + lm;
            const int mbase = m0 + waveM * 64 + mt * 16 + (lane >> 4) * 8;
            const float bv  = bias[n];
#pragma unroll
            for (int r = 0; r < 8; ++r) {
                C[(size_t)(mbase + r) * N_DIM + n] = acc[mt][nt][r] + bv;
            }
        }
    }
}

#else  // !HAVE_ASYNC_LDS — proven register-staging fallback

__global__ __launch_bounds__(256) void wmma_gemm_f16_fallback(
    const _Float16* __restrict__ A,
    const _Float16* __restrict__ B,
    const float*    __restrict__ bias,
    float*          __restrict__ C)
{
    __shared__ __align__(16) _Float16 ldsA[BM * LDS_STRIDE];
    __shared__ __align__(16) _Float16 ldsB[BN * LDS_STRIDE];

    const int tid   = threadIdx.x;
    const int lane  = tid & 31;
    const int wave  = tid >> 5;
    const int waveM = wave >> 2;
    const int waveN = wave & 3;

    const int m0 = blockIdx.y * BM;
    const int n0 = blockIdx.x * BN;

    const int lr = tid >> 2;
    const int lc = (tid & 3) * 8;

    const _Float16* gA = A + (size_t)(m0 + lr) * K_DIM + lc;
    const _Float16* gB = B + (size_t)(n0 + lr) * K_DIM + lc;

    v8h ra0 = *(const v8h*)(gA);
    v8h ra1 = *(const v8h*)(gA + (size_t)64 * K_DIM);
    v8h rb0 = *(const v8h*)(gB);
    v8h rb1 = *(const v8h*)(gB + (size_t)64 * K_DIM);

    v8f acc[4][2] = {};

    const int lm = lane & 15;
    const int kh = (lane >> 4) * 8;

    constexpr int KT = K_DIM / BK;
    for (int kt = 0; kt < KT; ++kt) {
        __syncthreads();
        *(v8h*)&ldsA[lr * LDS_STRIDE + lc]        = ra0;
        *(v8h*)&ldsA[(64 + lr) * LDS_STRIDE + lc] = ra1;
        *(v8h*)&ldsB[lr * LDS_STRIDE + lc]        = rb0;
        *(v8h*)&ldsB[(64 + lr) * LDS_STRIDE + lc] = rb1;
        __syncthreads();

        if (kt + 1 < KT) {
            const _Float16* gA2 = gA + (size_t)(kt + 1) * BK;
            const _Float16* gB2 = gB + (size_t)(kt + 1) * BK;
            ra0 = *(const v8h*)(gA2);
            ra1 = *(const v8h*)(gA2 + (size_t)64 * K_DIM);
            rb0 = *(const v8h*)(gB2);
            rb1 = *(const v8h*)(gB2 + (size_t)64 * K_DIM);
        }

        Frag16 fb[2];
#pragma unroll
        for (int nt = 0; nt < 2; ++nt) {
            const int row = waveN * 32 + nt * 16 + lm;
            fb[nt].h[0] = *(const v8h*)&ldsB[row * LDS_STRIDE + kh];
            fb[nt].h[1] = *(const v8h*)&ldsB[row * LDS_STRIDE + 16 + kh];
        }
        Frag16 fa[4];
#pragma unroll
        for (int mt = 0; mt < 4; ++mt) {
            const int row = waveM * 64 + mt * 16 + lm;
            fa[mt].h[0] = *(const v8h*)&ldsA[row * LDS_STRIDE + kh];
            fa[mt].h[1] = *(const v8h*)&ldsA[row * LDS_STRIDE + 16 + kh];
        }
#pragma unroll
        for (int mt = 0; mt < 4; ++mt) {
#pragma unroll
            for (int nt = 0; nt < 2; ++nt) {
                acc[mt][nt] = __builtin_amdgcn_wmma_f32_16x16x32_f16(
                    false, fa[mt].v, false, fb[nt].v,
                    (short)0, acc[mt][nt], false, false);
            }
        }
    }

#pragma unroll
    for (int mt = 0; mt < 4; ++mt) {
#pragma unroll
        for (int nt = 0; nt < 2; ++nt) {
            const int n     = n0 + waveN * 32 + nt * 16 + lm;
            const int mbase = m0 + waveM * 64 + mt * 16 + (lane >> 4) * 8;
            const float bv  = bias[n];
#pragma unroll
            for (int r = 0; r < 8; ++r) {
                C[(size_t)(mbase + r) * N_DIM + n] = acc[mt][nt][r] + bv;
            }
        }
    }
}
#endif

// ---------------- launch ----------------

extern "C" void kernel_launch(void* const* d_in, const int* in_sizes, int n_in,
                              void* d_out, int out_size, void* d_ws, size_t ws_size,
                              hipStream_t stream) {
    const float* x      = (const float*)d_in[0];
    const int*   wq     = (const int*)d_in[1];
    const float* absmax = (const float*)d_in[2];
    const float* code   = (const float*)d_in[3];
    const float* bias   = (const float*)d_in[4];
    float* out = (float*)d_out;

    _Float16* x16 = (_Float16*)d_ws;                       // 64 MiB
    _Float16* w16 = x16 + (size_t)M_DIM * K_DIM;           // +32 MiB

    const size_t nx = (size_t)M_DIM * K_DIM;
    const size_t nw = (size_t)N_DIM * K_DIM;

    convert_x_f16<<<(int)(nx / 4 / 256), 256, 0, stream>>>(x, x16);
    dequant_w_f16<<<(int)(nw / 4 / 256), 256, 0, stream>>>(wq, absmax, code, w16);

    dim3 grid(N_DIM / BN, M_DIM / BM);   // 32 x 64
#if HAVE_ASYNC_LDS
    wmma_gemm_f16_async<<<grid, 256, 0, stream>>>(x16, w16, bias, out);
#else
    wmma_gemm_f16_fallback<<<grid, 256, 0, stream>>>(x16, w16, bias, out);
#endif
}
